// Block_41893111005254
// MI455X (gfx1250) — matrix operand order repaired and checked by reading, not compiled, not run on hardware
//
#include <hip/hip_runtime.h>
#include <hip/hip_bf16.h>

// ---------------------------------------------------------------------------
// CDNA5 (gfx1250, wave32) types / helpers
// ---------------------------------------------------------------------------
typedef __bf16 bf16_t;
typedef __attribute__((ext_vector_type(16))) __bf16 v16bf;
typedef __attribute__((ext_vector_type(8)))  float  v8f;

#define WMMA_BF16(a, b, c) \
    __builtin_amdgcn_wmma_f32_16x16x32_bf16(false, (a), false, (b), (short)0, (c), false, false)

// Async global->LDS copy (gfx1250), tracked by ASYNCcnt.
static __device__ __forceinline__ void async_b128(unsigned lds_byte_off, const void* gaddr) {
    asm volatile("global_load_async_to_lds_b128 %0, %1, off"
                 :: "v"(lds_byte_off), "v"(gaddr) : "memory");
}
#define ASYNC_WAIT_0() asm volatile("s_wait_asynccnt 0x0" ::: "memory")
#define ASYNC_WAIT_4() asm volatile("s_wait_asynccnt 0x4" ::: "memory")

static __device__ __forceinline__ bf16_t f2bf(float f) { return (bf16_t)f; }

static __device__ __forceinline__ unsigned lds_off(const void* p) {
    return (unsigned)(size_t)p;   // AS(3) pointers are LDS byte offsets
}

static __device__ __forceinline__ v8f vzero8() {
    v8f z;
#pragma unroll
    for (int e = 0; e < 8; ++e) z[e] = 0.0f;
    return z;
}

// A-fragment (16x32 bf16, MxK) from LDS [row][k], row stride `stride` (bf16).
// ISA 7.12.2: lanes 0-15 -> M=lane, K={0..7,16..23}; lanes 16-31 -> K+8.
static __device__ __forceinline__ v16bf frag_a(const bf16_t* base, int lane, int stride) {
    int r  = lane & 15;
    int h8 = (lane >> 4) * 8;
    const bf16_t* row = base + r * stride;
    v16bf f;
#pragma unroll
    for (int j = 0; j < 8; ++j) {
        int k = ((j < 4) ? (2 * j) : (2 * j + 8)) + h8;
        f[2 * j]     = row[k];
        f[2 * j + 1] = row[k + 1];
    }
    return f;
}

// B-fragment (32x16 bf16, KxN) from LDS [n][k] (B transposed), stride `stride`.
// lanes 0-15 -> N=lane, K={0..7,16..23}; lanes 16-31 -> K+8.
static __device__ __forceinline__ v16bf frag_b(const bf16_t* base, int lane, int stride) {
    int n   = lane & 15;
    int h8  = (lane >> 4) * 8;
    const bf16_t* row = base + n * stride;
    v16bf f;
#pragma unroll
    for (int j = 0; j < 8; ++j) {
        int k = ((j < 4) ? (2 * j) : (2 * j + 8)) + h8;
        f[2 * j]     = row[k];
        f[2 * j + 1] = row[k + 1];
    }
    return f;
}

static __device__ __forceinline__ float gelu_exact(float x) {
    return 0.5f * x * (1.0f + erff(x * 0.7071067811865475f));
}

// ---------------------------------------------------------------------------
// fp32 -> bf16 flat conversion (for x)
// ---------------------------------------------------------------------------
__global__ __launch_bounds__(256)
void cvt_f32_bf16(const float* __restrict__ in, bf16_t* __restrict__ out) {
    size_t base = (size_t)blockIdx.x * 1024 + threadIdx.x * 4;
#pragma unroll
    for (int j = 0; j < 4; ++j) out[base + j] = f2bf(in[base + j]);
}

// ---------------------------------------------------------------------------
// fp32 [K,N] -> bf16 [N,K] transpose-convert (weights, once per launch)
// ---------------------------------------------------------------------------
__global__ __launch_bounds__(256)
void transpose_cvt(const float* __restrict__ in, bf16_t* __restrict__ out, int K, int N) {
    __shared__ float T[32][33];
    const int tid = threadIdx.x;
    const int n0 = blockIdx.x * 32;
    const int k0 = blockIdx.y * 32;
    {
        int r = tid >> 3, c = (tid & 7) * 4;
        const float* src = in + (size_t)(k0 + r) * N + n0 + c;
#pragma unroll
        for (int j = 0; j < 4; ++j) T[r][c + j] = src[j];
    }
    __syncthreads();
    {
        int n = tid >> 3, c = (tid & 7) * 4;
        bf16_t* dst = out + (size_t)(n0 + n) * K + k0 + c;
#pragma unroll
        for (int j = 0; j < 4; ++j) dst[j] = f2bf(T[c + j][n]);
    }
}

// ---------------------------------------------------------------------------
// WMMA GEMM: C = act(A[M,K]bf16 @ Bt[N,K]bf16^T + bias) (+ fp32 residual)
// flags: 1=GELU  2=+residual(fp32)  4=store bf16  8=store bf16 transposed [N,M]
// 128x128x32 tiles, 256 thr = 8 waves (2x4), async double-buffered LDS.
// ---------------------------------------------------------------------------
#define GBM 128
#define GBN 128
#define GBK 32
#define GSTR 40          // LDS row stride in bf16 (80B = 5*16B aligned chunks)
#define GBUF (GBM * GSTR)

__global__ __launch_bounds__(256)
void gemm_wmma_bf16(const bf16_t* __restrict__ A, const bf16_t* __restrict__ Bt,
                    const float* __restrict__ bias, const float* __restrict__ resid,
                    float* __restrict__ Cf, bf16_t* __restrict__ Cb,
                    int M, int N, int K, int flags) {
    __shared__ bf16_t As[2 * GBUF];
    __shared__ bf16_t Bs[2 * GBUF];

    const int tid  = threadIdx.x;
    const int lane = tid & 31;
    const int wave = tid >> 5;
    const int wm = wave >> 2;   // 0..1
    const int wn = wave & 3;    // 0..3

    const int rowBlock = blockIdx.y * GBM;
    const int colBlock = blockIdx.x * GBN;

    const unsigned asB = lds_off(&As[0]);
    const unsigned bsB = lds_off(&Bs[0]);

    v8f acc[4][2];
#pragma unroll
    for (int i = 0; i < 4; ++i)
#pragma unroll
        for (int j = 0; j < 2; ++j) acc[i][j] = vzero8();

    // Stage one 128x32 A tile + 128x32 Bt tile into buffer `buf` (async).
    auto stage = [&](int buf, int k0) {
        unsigned ab = asB + (unsigned)buf * (GBUF * 2);
        unsigned bb = bsB + (unsigned)buf * (GBUF * 2);
#pragma unroll
        for (int i = 0; i < 2; ++i) {
            int c = tid + i * 256;       // chunk 0..511
            int r = c >> 2, off = c & 3; // row, 16B-chunk within 64B row
            async_b128(ab + r * (GSTR * 2) + off * 16,
                       A  + (size_t)(rowBlock + r) * K + k0 + off * 8);
            async_b128(bb + r * (GSTR * 2) + off * 16,
                       Bt + (size_t)(colBlock + r) * K + k0 + off * 8);
        }
    };

    const int ntiles = K / GBK;
    stage(0, 0);
    for (int t = 0; t < ntiles; ++t) {
        const int cur = t & 1;
        if (t + 1 < ntiles) { stage(cur ^ 1, (t + 1) * GBK); ASYNC_WAIT_4(); }
        else                { ASYNC_WAIT_0(); }
        __syncthreads();

        const bf16_t* Ac = As + cur * GBUF;
        const bf16_t* Bc = Bs + cur * GBUF;
        v16bf afr[4], bfr[2];
#pragma unroll
        for (int i = 0; i < 4; ++i)
            afr[i] = frag_a(&Ac[(wm * 64 + i * 16) * GSTR], lane, GSTR);
#pragma unroll
        for (int j = 0; j < 2; ++j)
            bfr[j] = frag_b(&Bc[(wn * 32 + j * 16) * GSTR], lane, GSTR);
#pragma unroll
        for (int i = 0; i < 4; ++i)
#pragma unroll
            for (int j = 0; j < 2; ++j)
                acc[i][j] = WMMA_BF16(afr[i], bfr[j], acc[i][j]);
        __syncthreads();
    }

    // Epilogue. C layout: VGPR v -> row v + 8*(lane>=16), col = lane&15.
    const int h  = lane >> 4;
    const int nc = lane & 15;
#pragma unroll
    for (int i = 0; i < 4; ++i) {
#pragma unroll
        for (int j = 0; j < 2; ++j) {
            int col = colBlock + wn * 32 + j * 16 + nc;
            float bv = bias[col];
#pragma unroll
            for (int v = 0; v < 8; ++v) {
                int row = rowBlock + wm * 64 + i * 16 + v + 8 * h;
                float val = acc[i][j][v] + bv;
                if (flags & 1) val = gelu_exact(val);
                if (flags & 2) val += resid[(size_t)row * N + col];
                if (flags & 8)      Cb[(size_t)col * M + row] = f2bf(val); // transposed bf16
                else if (flags & 4) Cb[(size_t)row * N + col] = f2bf(val); // bf16
                else                Cf[(size_t)row * N + col] = val;        // fp32
            }
        }
    }
}

// ---------------------------------------------------------------------------
// Flash attention (no 1/sqrt(d) scale, matching reference).
// Q,K bf16 [B*S,1024] head-major cols; Vt bf16 [1024, B*S] (pre-transposed).
// Output bf16 [B*S,1024]. Grid (S/128, H, B); 8 waves, 16 query rows each.
// ---------------------------------------------------------------------------
#define AQ   128
#define AKV  64
#define AHD  64
#define ASTR 72     // LDS row stride in bf16 (144B = 9*16B)

__global__ __launch_bounds__(256)
void flash_attn_wmma(const bf16_t* __restrict__ Q, const bf16_t* __restrict__ K,
                     const bf16_t* __restrict__ Vt, bf16_t* __restrict__ O,
                     int S, int Mtot) {
    __shared__ bf16_t Qs[AQ  * ASTR];   // [qrow][d]
    __shared__ bf16_t Ks[AKV * ASTR];   // [krow][d]
    __shared__ bf16_t Vs[AHD * ASTR];   // [d][ktok]
    __shared__ bf16_t Ps[AQ  * ASTR];   // [qrow][ktok] (per-wave private rows)

    const int tid  = threadIdx.x;
    const int lane = tid & 31;
    const int wave = tid >> 5;
    const int b  = blockIdx.z;
    const int hh = blockIdx.y;
    const int q0 = blockIdx.x * AQ;
    const int D  = 1024;

    const unsigned qsB = lds_off(&Qs[0]);
    const unsigned ksB = lds_off(&Ks[0]);
    const unsigned vsB = lds_off(&Vs[0]);

    // Stage Q tile (128x64 bf16 = 1024 16B-chunks) asynchronously, once.
#pragma unroll
    for (int i = 0; i < 4; ++i) {
        int c = tid + i * 256;
        int r = c >> 3, off = c & 7;
        async_b128(qsB + r * (ASTR * 2) + off * 16,
                   Q + (size_t)(b * S + q0 + r) * D + hh * AHD + off * 8);
    }

    float m_run[8], l_run[8];
    v8f acc_o[4];
#pragma unroll
    for (int i = 0; i < 8; ++i) { m_run[i] = -3.0e38f; l_run[i] = 0.0f; }
#pragma unroll
    for (int nt = 0; nt < 4; ++nt) acc_o[nt] = vzero8();

    const int hl = lane >> 4;
    const int nc = lane & 15;

    for (int kv0 = 0; kv0 < S; kv0 += AKV) {
        __syncthreads();   // previous K/V readers done
        // K tile 64x64 and Vt tile 64x64: 512 chunks each -> 2+2 per thread
#pragma unroll
        for (int i = 0; i < 2; ++i) {
            int c = tid + i * 256;
            int r = c >> 3, off = c & 7;
            async_b128(ksB + r * (ASTR * 2) + off * 16,
                       K  + (size_t)(b * S + kv0 + r) * D + hh * AHD + off * 8);
            async_b128(vsB + r * (ASTR * 2) + off * 16,
                       Vt + (size_t)(hh * AHD + r) * Mtot + b * S + kv0 + off * 8);
        }
        ASYNC_WAIT_0();
        __syncthreads();

        // S_w(16x64) = Q_w @ K^T ; K-dim 64 = 2 WMMA steps.
        v8f sc[4];
#pragma unroll
        for (int nt = 0; nt < 4; ++nt) sc[nt] = vzero8();
#pragma unroll
        for (int ks = 0; ks < 2; ++ks) {
            v16bf a = frag_a(&Qs[(wave * 16) * ASTR + ks * 32], lane, ASTR);
#pragma unroll
            for (int nt = 0; nt < 4; ++nt) {
                v16bf bf = frag_b(&Ks[(nt * 16) * ASTR + ks * 32], lane, ASTR);
                sc[nt] = WMMA_BF16(a, bf, sc[nt]);
            }
        }

        // Online softmax; row r = i + 8*hl spans the 16 lanes of a half-group.
#pragma unroll
        for (int i = 0; i < 8; ++i) {
            float rmax = -3.0e38f;
#pragma unroll
            for (int nt = 0; nt < 4; ++nt) rmax = fmaxf(rmax, sc[nt][i]);
#pragma unroll
            for (int mk = 8; mk >= 1; mk >>= 1) rmax = fmaxf(rmax, __shfl_xor(rmax, mk, 32));
            float mnew  = fmaxf(m_run[i], rmax);
            float scale = __expf(m_run[i] - mnew);
            float rsum = 0.0f;
#pragma unroll
            for (int nt = 0; nt < 4; ++nt) {
                float p = __expf(sc[nt][i] - mnew);
                sc[nt][i] = p;
                rsum += p;
            }
#pragma unroll
            for (int mk = 8; mk >= 1; mk >>= 1) rsum += __shfl_xor(rsum, mk, 32);
            l_run[i] = l_run[i] * scale + rsum;
            m_run[i] = mnew;
#pragma unroll
            for (int nt = 0; nt < 4; ++nt) acc_o[nt][i] *= scale;
        }

        // Stage P (16x64 bf16) into this wave's private LDS rows.
#pragma unroll
        for (int nt = 0; nt < 4; ++nt)
#pragma unroll
            for (int i = 0; i < 8; ++i)
                Ps[(wave * 16 + i + 8 * hl) * ASTR + nt * 16 + nc] = f2bf(sc[nt][i]);

        // O_w += P_w(16x64) @ V(64x64).
#pragma unroll
        for (int ks = 0; ks < 2; ++ks) {
            v16bf a = frag_a(&Ps[(wave * 16) * ASTR + ks * 32], lane, ASTR);
#pragma unroll
            for (int nt = 0; nt < 4; ++nt) {
                v16bf bf = frag_b(&Vs[(nt * 16) * ASTR + ks * 32], lane, ASTR);
                acc_o[nt] = WMMA_BF16(a, bf, acc_o[nt]);
            }
        }
    }

    // Normalize and write bf16.
#pragma unroll
    for (int nt = 0; nt < 4; ++nt) {
#pragma unroll
        for (int i = 0; i < 8; ++i) {
            int row = q0 + wave * 16 + i + 8 * hl;
            int col = hh * AHD + nt * 16 + nc;
            O[(size_t)(b * S + row) * D + col] = f2bf(acc_o[nt][i] / l_run[i]);
        }
    }
}

// ---------------------------------------------------------------------------
// LayerNorm (N=1024, biased variance). Optional bf16 secondary output.
// ---------------------------------------------------------------------------
__global__ __launch_bounds__(256)
void layernorm_rows(const float* __restrict__ X, const float* __restrict__ g,
                    const float* __restrict__ beta, float* __restrict__ Y,
                    bf16_t* __restrict__ Yb, int N) {
    __shared__ float red_s[8];
    __shared__ float red_q[8];
    const int row  = blockIdx.x;
    const int tid  = threadIdx.x;
    const int lane = tid & 31;
    const int wave = tid >> 5;
    const float* x = X + (size_t)row * N;

    float v[4];
    float s = 0.0f, q = 0.0f;
#pragma unroll
    for (int i = 0; i < 4; ++i) {
        v[i] = x[tid + i * 256];
        s += v[i];
        q += v[i] * v[i];
    }
#pragma unroll
    for (int mk = 16; mk >= 1; mk >>= 1) {
        s += __shfl_xor(s, mk, 32);
        q += __shfl_xor(q, mk, 32);
    }
    if (lane == 0) { red_s[wave] = s; red_q[wave] = q; }
    __syncthreads();
    s = red_s[lane & 7];
    q = red_q[lane & 7];
#pragma unroll
    for (int mk = 4; mk >= 1; mk >>= 1) {
        s += __shfl_xor(s, mk, 32);
        q += __shfl_xor(q, mk, 32);
    }
    __syncthreads();
    if (tid == 0) { red_s[0] = s; red_q[0] = q; }
    __syncthreads();
    s = red_s[0];
    q = red_q[0];

    const float inv_n = 1.0f / (float)N;
    const float mu = s * inv_n;
    const float var = q * inv_n - mu * mu;
    const float rstd = rsqrtf(var + 1e-5f);
    float* y = Y + (size_t)row * N;
#pragma unroll
    for (int i = 0; i < 4; ++i) {
        int c = tid + i * 256;
        float o = (v[i] - mu) * rstd * g[c] + beta[c];
        y[c] = o;
        if (Yb) Yb[(size_t)row * N + c] = f2bf(o);
    }
}

// ---------------------------------------------------------------------------
// Host launcher
// ---------------------------------------------------------------------------
extern "C" void kernel_launch(void* const* d_in, const int* in_sizes, int n_in,
                              void* d_out, int out_size, void* d_ws, size_t ws_size,
                              hipStream_t stream) {
    (void)in_sizes; (void)n_in; (void)out_size; (void)ws_size;

    const int Bb = 2, S = 2048, D = 1024, FF = 4096;
    const int Mrows = Bb * S;  // 4096

    const float* x   = (const float*)d_in[0];
    const float* Wq  = (const float*)d_in[1];
    const float* bq  = (const float*)d_in[2];
    const float* Wk  = (const float*)d_in[3];
    const float* bk  = (const float*)d_in[4];
    const float* Wv  = (const float*)d_in[5];
    const float* bv  = (const float*)d_in[6];
    const float* Wo  = (const float*)d_in[7];
    const float* bo  = (const float*)d_in[8];
    const float* W1  = (const float*)d_in[9];
    const float* b1  = (const float*)d_in[10];
    const float* W2  = (const float*)d_in[11];
    const float* b2  = (const float*)d_in[12];
    const float* g1  = (const float*)d_in[13];
    const float* be1 = (const float*)d_in[14];
    const float* g2  = (const float*)d_in[15];
    const float* be2 = (const float*)d_in[16];

    // Workspace layout (bytes)
    char* ws = (char*)d_ws;
    const size_t MD2  = (size_t)Mrows * D * 2;      // bf16 [M,D]   = 8MB
    const size_t DD2  = (size_t)D * D * 2;          // bf16 [D,D]   = 2MB
    const size_t DF2  = (size_t)D * FF * 2;         // bf16 weights = 8MB
    const size_t MD4  = (size_t)Mrows * D * 4;      // fp32 [M,D]   = 16MB
    const size_t MF2  = (size_t)Mrows * FF * 2;     // bf16 [M,FF]  = 32MB

    size_t o = 0;
    bf16_t* xb  = (bf16_t*)(ws + o); o += MD2;
    bf16_t* wqb = (bf16_t*)(ws + o); o += DD2;
    bf16_t* wkb = (bf16_t*)(ws + o); o += DD2;
    bf16_t* wvb = (bf16_t*)(ws + o); o += DD2;
    bf16_t* wob = (bf16_t*)(ws + o); o += DD2;
    bf16_t* w1b = (bf16_t*)(ws + o); o += DF2;
    bf16_t* w2b = (bf16_t*)(ws + o); o += DF2;
    bf16_t* qbf = (bf16_t*)(ws + o); o += MD2;      // overlaid later by y1
    bf16_t* kbf = (bf16_t*)(ws + o); o += MD2;
    bf16_t* vtb = (bf16_t*)(ws + o); o += MD2;
    bf16_t* atb = (bf16_t*)(ws + o); o += MD2;      // attn out; reused as x1b
    float*  x1f = (float* )(ws + o); o += MD4;
    float*  y0  = (float* )(ws + o); o += MD4;
    bf16_t* hb  = (bf16_t*)(ws + o); o += MF2;
    float*  y1  = (float*)qbf;                      // qbf+kbf dead after attention
    bf16_t* x1b = atb;                              // atb dead after Wo GEMM

    dim3 blk(256);

    // ---- One-time (per launch) bf16 conversions -------------------------
    cvt_f32_bf16<<<dim3((Mrows * D) / 1024), blk, 0, stream>>>(x, xb);
    transpose_cvt<<<dim3(D / 32,  D / 32),  blk, 0, stream>>>(Wq, wqb, D, D);
    transpose_cvt<<<dim3(D / 32,  D / 32),  blk, 0, stream>>>(Wk, wkb, D, D);
    transpose_cvt<<<dim3(D / 32,  D / 32),  blk, 0, stream>>>(Wv, wvb, D, D);
    transpose_cvt<<<dim3(D / 32,  D / 32),  blk, 0, stream>>>(Wo, wob, D, D);
    transpose_cvt<<<dim3(FF / 32, D / 32),  blk, 0, stream>>>(W1, w1b, D, FF);
    transpose_cvt<<<dim3(D / 32,  FF / 32), blk, 0, stream>>>(W2, w2b, FF, D);

    dim3 gD (D  / GBN, Mrows / GBM);   // (8, 32)
    dim3 gFF(FF / GBN, Mrows / GBM);   // (32, 32)

    // ---- QKV (bf16 outputs; V stored transposed) ------------------------
    gemm_wmma_bf16<<<gD, blk, 0, stream>>>(xb, wqb, bq, nullptr, nullptr, qbf, Mrows, D, D, 4);
    gemm_wmma_bf16<<<gD, blk, 0, stream>>>(xb, wkb, bk, nullptr, nullptr, kbf, Mrows, D, D, 4);
    gemm_wmma_bf16<<<gD, blk, 0, stream>>>(xb, wvb, bv, nullptr, nullptr, vtb, Mrows, D, D, 8);

    // ---- Flash attention -------------------------------------------------
    flash_attn_wmma<<<dim3(S / AQ, 16, Bb), blk, 0, stream>>>(qbf, kbf, vtb, atb, S, Mrows);

    // ---- Output projection + residual (fp32 out) ------------------------
    gemm_wmma_bf16<<<gD, blk, 0, stream>>>(atb, wob, bo, x, y0, nullptr, Mrows, D, D, 2);

    // ---- LN1 (fp32 + bf16 outputs) ---------------------------------------
    layernorm_rows<<<dim3(Mrows), blk, 0, stream>>>(y0, g1, be1, x1f, x1b, D);

    // ---- FFN1: gelu(x1 @ W1 + b1) -> bf16 --------------------------------
    gemm_wmma_bf16<<<gFF, blk, 0, stream>>>(x1b, w1b, b1, nullptr, nullptr, hb, Mrows, FF, D, 1 | 4);

    // ---- FFN2: x1 + h @ W2 + b2 -> fp32 ----------------------------------
    gemm_wmma_bf16<<<gD, blk, 0, stream>>>(hb, w2b, b2, x1f, y1, nullptr, Mrows, D, FF, 2);

    // ---- LN2 -> output ----------------------------------------------------
    layernorm_rows<<<dim3(Mrows), blk, 0, stream>>>(y1, g2, be2, (float*)d_out, nullptr, D);
}
